// MoELayer_43525198578337
// MI455X (gfx1250) — matrix operand order repaired and checked by reading, not compiled, not run on hardware
//
#include <hip/hip_runtime.h>
#include <hip/hip_bf16.h>
#include <stdint.h>

#define NTOK 8192
#define CDIM 1024
#define HDIM 2048
#define ENUM 8

// ---------------------------------------------------------------------------
// CDNA5 async load-to-LDS support (guarded: falls back to register staging).
// Builtin signature (confirmed by compiler diagnostics):
//   void __builtin_amdgcn_global_load_async_to_lds_b128(
//       v4i addrspace(1)* src, v4i addrspace(3)* dst, imm int off, imm int cpol)
// ---------------------------------------------------------------------------
#if defined(__AMDGCN__) && __has_builtin(__builtin_amdgcn_global_load_async_to_lds_b128)
#define HAVE_ASYNC_LDS 1
typedef int v4i_async __attribute__((vector_size(16)));
typedef __attribute__((address_space(1))) v4i_async as1_v4i;
typedef __attribute__((address_space(3))) v4i_async as3_v4i;
#define ASYNC_CP_B128(gp, lp)                                        \
  __builtin_amdgcn_global_load_async_to_lds_b128((as1_v4i*)(gp),     \
                                                 (as3_v4i*)(lp), 0, 0)
#else
#define HAVE_ASYNC_LDS 0
#endif

__device__ __forceinline__ void wait_async0() {
#if HAVE_ASYNC_LDS
#if __has_builtin(__builtin_amdgcn_s_wait_asynccnt)
  __builtin_amdgcn_s_wait_asynccnt(0);
#else
  asm volatile("s_wait_asynccnt 0" ::: "memory");
#endif
#endif
}

typedef __attribute__((ext_vector_type(16))) __bf16 v16bf;
typedef __attribute__((ext_vector_type(8)))  float  v8f;

union FragU { uint4 q[2]; v16bf v; __bf16 b[16]; };

// A-fragment (16x32 bf16): lane<16 row=lane K=0..15; lane>=16 row=lane-16
// K=16..31. LDS tile row-major [M][32] -> two contiguous 16B loads.
__device__ __forceinline__ v16bf lds_a_frag(const __bf16* At, int mrow, int lane) {
  const int half = lane >> 4, r = lane & 15;
  const __bf16* base = At + (size_t)(mrow + r) * 32 + half * 16;
  FragU f;
  f.q[0] = *(const uint4*)(base);
  f.q[1] = *(const uint4*)(base + 8);
  return f.v;
}

// B-fragment (32x16 bf16): lane<16 col=lane K=0..15; lane>=16 col=lane-16
// K=16..31. LDS tile stored [N][32] (K-contiguous) -> 32 contiguous bytes.
__device__ __forceinline__ v16bf lds_b_frag(const __bf16* Bt, int ncol, int lane) {
  const int half = lane >> 4, n = lane & 15;
  const __bf16* base = Bt + (size_t)(ncol + n) * 32 + half * 16;
  FragU f;
  f.q[0] = *(const uint4*)(base);
  f.q[1] = *(const uint4*)(base + 8);
  return f.v;
}

__device__ __forceinline__ v8f wmma_bf16(v16bf a, v16bf b, v8f c) {
  return __builtin_amdgcn_wmma_f32_16x16x32_bf16(false, a, false, b, (short)0, c,
                                                 false, false);
}

__device__ __forceinline__ float silu_f(float x) { return x / (1.0f + __expf(-x)); }

// ---------------------------------------------------------------------------
// fp32 [K][N] -> bf16 [N][K] transpose+convert (32x32 tiles via LDS).
// blockIdx.z selects matrix (for stacked expert weights).
// ---------------------------------------------------------------------------
__global__ __launch_bounds__(256) void moe_cvt_t(const float* __restrict__ s,
                                                 __bf16* __restrict__ d,
                                                 int K, int N) {
  __shared__ __bf16 tile[32][33];
  const size_t base = (size_t)blockIdx.z * K * N;
  const int k0 = blockIdx.y * 32, n0 = blockIdx.x * 32;
  const int tx = threadIdx.x & 31, ty = threadIdx.x >> 5;
#pragma unroll
  for (int j = 0; j < 4; ++j)
    tile[ty + j * 8][tx] = (__bf16)s[base + (size_t)(k0 + ty + j * 8) * N + n0 + tx];
  __syncthreads();
  const int row = threadIdx.x >> 3;       // n within tile
  const int cs  = (threadIdx.x & 7) * 4;  // k within tile (4 elems -> 8B store)
  union { uint2 u; __bf16 b[4]; } v;
#pragma unroll
  for (int j = 0; j < 4; ++j) v.b[j] = tile[cs + j][row];
  *(uint2*)(d + base + (size_t)(n0 + row) * K + k0 + cs) = v.u;
}

// ---------------------------------------------------------------------------
// Router: one wave per token. Logits (output #2), top-2 softmax, per-expert
// token list append, and x fp32->bf16 conversion.
// ---------------------------------------------------------------------------
__global__ __launch_bounds__(256) void moe_router(
    const float* __restrict__ x, const float* __restrict__ gw,
    __bf16* __restrict__ xb, float* __restrict__ logits,
    int* __restrict__ cnt, int* __restrict__ lidx, float* __restrict__ lw) {
  const int lane = threadIdx.x & 31;
  const int t = blockIdx.x * 8 + (threadIdx.x >> 5);
  float acc[ENUM];
#pragma unroll
  for (int e = 0; e < ENUM; ++e) acc[e] = 0.f;
  const size_t xoff = (size_t)t * CDIM;
  for (int i = 0; i < CDIM / 32; ++i) {
    const int c = i * 32 + lane;
    const float xv = x[xoff + c];
    xb[xoff + c] = (__bf16)xv;
#pragma unroll
    for (int e = 0; e < ENUM; ++e) acc[e] += xv * gw[c * ENUM + e];
  }
#pragma unroll
  for (int e = 0; e < ENUM; ++e) {
#pragma unroll
    for (int off = 16; off > 0; off >>= 1) acc[e] += __shfl_xor(acc[e], off, 32);
  }
  if (lane == 0) {
#pragma unroll
    for (int e = 0; e < ENUM; ++e) logits[(size_t)t * ENUM + e] = acc[e];
    int e1 = 0;
    for (int e = 1; e < ENUM; ++e) if (acc[e] > acc[e1]) e1 = e;
    int e2 = (e1 == 0) ? 1 : 0;
    for (int e = 0; e < ENUM; ++e) if (e != e1 && acc[e] > acc[e2]) e2 = e;
    const float w1 = 1.f / (1.f + __expf(acc[e2] - acc[e1]));
    const float w2 = 1.f - w1;
    const int p1 = atomicAdd(&cnt[e1], 1);
    lidx[e1 * NTOK + p1] = t;
    lw[e1 * NTOK + p1] = w1;
    const int p2 = atomicAdd(&cnt[e2], 1);
    lidx[e2 * NTOK + p2] = t;
    lw[e2 * NTOK + p2] = w2;
  }
}

// ---------------------------------------------------------------------------
// Fused up-projection: h = silu(X*Wg) .* (X*Wu). Weights pre-transposed to
// [H][C]. 64x128 block tile, BK=32, LDS ping-pong; staging via CDNA5 async
// load-to-LDS (ASYNCcnt) when available, else pipelined register staging.
// ---------------------------------------------------------------------------
template <bool EXPERT>
__global__ __launch_bounds__(256) void moe_up(
    const __bf16* __restrict__ xb, const __bf16* __restrict__ wgT,
    const __bf16* __restrict__ wuT, __bf16* __restrict__ hb,
    const int* __restrict__ cnt, const int* __restrict__ lidx) {
  __shared__ __bf16 At[2][64 * 32];
  __shared__ __bf16 Bg[2][128 * 32];
  __shared__ __bf16 Bu[2][128 * 32];
  const int count = EXPERT ? *cnt : NTOK;
  const int m0 = blockIdx.y * 64;
  if (m0 >= count) return;
  const int n0 = blockIdx.x * 128;
  const int tid = threadIdx.x, lane = tid & 31, wave = tid >> 5;
  const int wm = wave >> 2, wn = wave & 3;

  // Staging assignments (constant over k-steps)
  const int ar = tid >> 2, ac = (tid & 3) * 8;   // A: 64 rows x 32, 16B/thread
  const int bn = tid >> 1, bk = (tid & 1) * 16;  // B: 128 rows x 32, 32B/thread
  int atok;
  if (EXPERT) {
    const int ridx = m0 + ar;
    // Out-of-range rows produce garbage D rows that are masked at the store;
    // clamp the gather index so the load is always in-bounds.
    atok = (ridx < count) ? lidx[ridx] : 0;
  } else {
    atok = m0 + ar;
  }
  const __bf16* aptr = xb + (size_t)atok * CDIM + ac;
  const __bf16* gptr = wgT + (size_t)(n0 + bn) * CDIM + bk;
  const __bf16* uptr = wuT + (size_t)(n0 + bn) * CDIM + bk;

#if HAVE_ASYNC_LDS
  auto astage = [&](int buf, int kt) {
    const int k0 = kt * 32;
    ASYNC_CP_B128(aptr + k0, &At[buf][ar * 32 + ac]);
    ASYNC_CP_B128(gptr + k0, &Bg[buf][bn * 32 + bk]);
    ASYNC_CP_B128(gptr + k0 + 8, &Bg[buf][bn * 32 + bk + 8]);
    ASYNC_CP_B128(uptr + k0, &Bu[buf][bn * 32 + bk]);
    ASYNC_CP_B128(uptr + k0 + 8, &Bu[buf][bn * 32 + bk + 8]);
  };
#else
  uint4 ra, rg0, rg1, ru0, ru1;
  auto gload = [&](int kt) {
    const int k0 = kt * 32;
    ra = *(const uint4*)(aptr + k0);
    const uint4* g = (const uint4*)(gptr + k0);
    const uint4* u = (const uint4*)(uptr + k0);
    rg0 = g[0]; rg1 = g[1];
    ru0 = u[0]; ru1 = u[1];
  };
  auto lstore = [&](int buf) {
    *(uint4*)(&At[buf][ar * 32 + ac]) = ra;
    uint4* g = (uint4*)(&Bg[buf][bn * 32 + bk]);
    g[0] = rg0; g[1] = rg1;
    uint4* u = (uint4*)(&Bu[buf][bn * 32 + bk]);
    u[0] = ru0; u[1] = ru1;
  };
#endif

  const v8f vz = {0.f, 0.f, 0.f, 0.f, 0.f, 0.f, 0.f, 0.f};
  v8f accG[2][2] = {{vz, vz}, {vz, vz}};
  v8f accU[2][2] = {{vz, vz}, {vz, vz}};

#if HAVE_ASYNC_LDS
  astage(0, 0);
  wait_async0();
#else
  gload(0);
  lstore(0);
#endif
  __syncthreads();
  const int KT = CDIM / 32;
  for (int kt = 0; kt < KT; ++kt) {
    const int cur = kt & 1;
#if HAVE_ASYNC_LDS
    if (kt + 1 < KT) astage(cur ^ 1, kt + 1);
#else
    if (kt + 1 < KT) gload(kt + 1);
#endif
    v16bf a[2], g2[2], u2[2];
#pragma unroll
    for (int mt = 0; mt < 2; ++mt) a[mt] = lds_a_frag(At[cur], wm * 32 + mt * 16, lane);
#pragma unroll
    for (int nt = 0; nt < 2; ++nt) {
      g2[nt] = lds_b_frag(Bg[cur], wn * 32 + nt * 16, lane);
      u2[nt] = lds_b_frag(Bu[cur], wn * 32 + nt * 16, lane);
    }
#pragma unroll
    for (int mt = 0; mt < 2; ++mt)
#pragma unroll
      for (int nt = 0; nt < 2; ++nt) {
        accG[mt][nt] = wmma_bf16(a[mt], g2[nt], accG[mt][nt]);
        accU[mt][nt] = wmma_bf16(a[mt], u2[nt], accU[mt][nt]);
      }
#if HAVE_ASYNC_LDS
    if (kt + 1 < KT) wait_async0();
#else
    if (kt + 1 < KT) lstore(cur ^ 1);
#endif
    __syncthreads();
  }

  const int hi = lane >> 4, nn = lane & 15;
#pragma unroll
  for (int mt = 0; mt < 2; ++mt)
#pragma unroll
    for (int nt = 0; nt < 2; ++nt)
#pragma unroll
      for (int i = 0; i < 8; ++i) {
        const int ridx = m0 + wm * 32 + mt * 16 + hi * 8 + i;
        const int col = n0 + wn * 32 + nt * 16 + nn;
        if (!EXPERT || ridx < count) {
          const float h = silu_f(accG[mt][nt][i]) * accU[mt][nt][i];
          hb[(size_t)ridx * HDIM + col] = (__bf16)h;
        }
      }
}

// ---------------------------------------------------------------------------
// Down-projection: out = H * Wd  (Wd pre-transposed to [C][H]).
// Shared: plain store; expert: weighted atomic scatter-add via token list.
// ---------------------------------------------------------------------------
template <bool EXPERT>
__global__ __launch_bounds__(256) void moe_down(
    const __bf16* __restrict__ hb, const __bf16* __restrict__ wdT,
    float* __restrict__ out, const int* __restrict__ cnt,
    const int* __restrict__ lidx, const float* __restrict__ lw) {
  __shared__ __bf16 At[2][64 * 32];
  __shared__ __bf16 Bt[2][128 * 32];
  const int count = EXPERT ? *cnt : NTOK;
  const int m0 = blockIdx.y * 64;
  if (m0 >= count) return;
  const int n0 = blockIdx.x * 128;
  const int tid = threadIdx.x, lane = tid & 31, wave = tid >> 5;
  const int wm = wave >> 2, wn = wave & 3;

  const int ar = tid >> 2, ac = (tid & 3) * 8;
  const int bn = tid >> 1, bk = (tid & 1) * 16;
  const __bf16* aptr = hb + (size_t)(m0 + ar) * HDIM + ac;
  const __bf16* bptr = wdT + (size_t)(n0 + bn) * HDIM + bk;

#if HAVE_ASYNC_LDS
  auto astage = [&](int buf, int kt) {
    const int k0 = kt * 32;
    ASYNC_CP_B128(aptr + k0, &At[buf][ar * 32 + ac]);
    ASYNC_CP_B128(bptr + k0, &Bt[buf][bn * 32 + bk]);
    ASYNC_CP_B128(bptr + k0 + 8, &Bt[buf][bn * 32 + bk + 8]);
  };
#else
  uint4 ra, rb0, rb1;
  auto gload = [&](int kt) {
    const int k0 = kt * 32;
    ra = *(const uint4*)(aptr + k0);
    const uint4* b = (const uint4*)(bptr + k0);
    rb0 = b[0]; rb1 = b[1];
  };
  auto lstore = [&](int buf) {
    *(uint4*)(&At[buf][ar * 32 + ac]) = ra;
    uint4* b = (uint4*)(&Bt[buf][bn * 32 + bk]);
    b[0] = rb0; b[1] = rb1;
  };
#endif

  const v8f vz = {0.f, 0.f, 0.f, 0.f, 0.f, 0.f, 0.f, 0.f};
  v8f acc[2][2] = {{vz, vz}, {vz, vz}};

#if HAVE_ASYNC_LDS
  astage(0, 0);
  wait_async0();
#else
  gload(0);
  lstore(0);
#endif
  __syncthreads();
  const int KT = HDIM / 32;
  for (int kt = 0; kt < KT; ++kt) {
    const int cur = kt & 1;
#if HAVE_ASYNC_LDS
    if (kt + 1 < KT) astage(cur ^ 1, kt + 1);
#else
    if (kt + 1 < KT) gload(kt + 1);
#endif
    v16bf a[2], b2[2];
#pragma unroll
    for (int mt = 0; mt < 2; ++mt) a[mt] = lds_a_frag(At[cur], wm * 32 + mt * 16, lane);
#pragma unroll
    for (int nt = 0; nt < 2; ++nt) b2[nt] = lds_b_frag(Bt[cur], wn * 32 + nt * 16, lane);
#pragma unroll
    for (int mt = 0; mt < 2; ++mt)
#pragma unroll
      for (int nt = 0; nt < 2; ++nt)
        acc[mt][nt] = wmma_bf16(a[mt], b2[nt], acc[mt][nt]);
#if HAVE_ASYNC_LDS
    if (kt + 1 < KT) wait_async0();
#else
    if (kt + 1 < KT) lstore(cur ^ 1);
#endif
    __syncthreads();
  }

  const int hi = lane >> 4, nn = lane & 15;
#pragma unroll
  for (int mt = 0; mt < 2; ++mt)
#pragma unroll
    for (int nt = 0; nt < 2; ++nt)
#pragma unroll
      for (int i = 0; i < 8; ++i) {
        const int ridx = m0 + wm * 32 + mt * 16 + hi * 8 + i;
        const int col = n0 + wn * 32 + nt * 16 + nn;
        if (EXPERT) {
          if (ridx < count) {
            const int tok = lidx[ridx];
            atomicAdd(out + (size_t)tok * CDIM + col, lw[ridx] * acc[mt][nt][i]);
          }
        } else {
          out[(size_t)ridx * CDIM + col] = acc[mt][nt][i];
        }
      }
}

// ---------------------------------------------------------------------------
extern "C" void kernel_launch(void* const* d_in, const int* in_sizes, int n_in,
                              void* d_out, int out_size, void* d_ws, size_t ws_size,
                              hipStream_t stream) {
  const float* x  = (const float*)d_in[0];
  const float* gw = (const float*)d_in[1];
  const float* sg = (const float*)d_in[2];
  const float* su = (const float*)d_in[3];
  const float* sd = (const float*)d_in[4];
  const float* eg = (const float*)d_in[5];
  const float* eu = (const float*)d_in[6];
  const float* ed = (const float*)d_in[7];
  float* out = (float*)d_out;
  float* logits = out + (size_t)NTOK * CDIM;

  char* ws = (char*)d_ws;
  size_t off = 0;
  auto take = [&](size_t bytes) -> char* {
    char* p = ws + off;
    off = (off + bytes + 255) & ~(size_t)255;
    return p;
  };
  __bf16* xb  = (__bf16*)take((size_t)NTOK * CDIM * 2);
  __bf16* hb  = (__bf16*)take((size_t)NTOK * HDIM * 2);
  __bf16* wsg = (__bf16*)take((size_t)CDIM * HDIM * 2);   // [H][C]
  __bf16* wsu = (__bf16*)take((size_t)CDIM * HDIM * 2);   // [H][C]
  __bf16* wsd = (__bf16*)take((size_t)HDIM * CDIM * 2);   // [C][H]
  __bf16* weg = (__bf16*)take((size_t)ENUM * CDIM * HDIM * 2);
  __bf16* weu = (__bf16*)take((size_t)ENUM * CDIM * HDIM * 2);
  __bf16* wed = (__bf16*)take((size_t)ENUM * HDIM * CDIM * 2);
  int*   cnt  = (int*)take(ENUM * sizeof(int));
  int*   lidx = (int*)take((size_t)ENUM * NTOK * sizeof(int));
  float* lw   = (float*)take((size_t)ENUM * NTOK * sizeof(float));

  (void)hipMemsetAsync(cnt, 0, ENUM * sizeof(int), stream);

  const int CH = CDIM * HDIM;
  // Transpose+convert: (K x N) fp32 -> (N x K) bf16
  dim3 tCH(HDIM / 32, CDIM / 32, 1);   // K=C, N=H
  dim3 tHC(CDIM / 32, HDIM / 32, 1);   // K=H, N=C
  dim3 tCHe(HDIM / 32, CDIM / 32, ENUM);
  dim3 tHCe(CDIM / 32, HDIM / 32, ENUM);
  moe_cvt_t<<<tCH, 256, 0, stream>>>(sg, wsg, CDIM, HDIM);
  moe_cvt_t<<<tCH, 256, 0, stream>>>(su, wsu, CDIM, HDIM);
  moe_cvt_t<<<tHC, 256, 0, stream>>>(sd, wsd, HDIM, CDIM);
  moe_cvt_t<<<tCHe, 256, 0, stream>>>(eg, weg, CDIM, HDIM);
  moe_cvt_t<<<tCHe, 256, 0, stream>>>(eu, weu, CDIM, HDIM);
  moe_cvt_t<<<tHCe, 256, 0, stream>>>(ed, wed, HDIM, CDIM);

  moe_router<<<NTOK / 8, 256, 0, stream>>>(x, gw, xb, logits, cnt, lidx, lw);

  dim3 upg(HDIM / 128, NTOK / 64);
  dim3 dng(CDIM / 128, NTOK / 64);
  moe_up<false><<<upg, 256, 0, stream>>>(xb, wsg, wsu, hb, nullptr, nullptr);
  moe_down<false><<<dng, 256, 0, stream>>>(hb, wsd, out, nullptr, nullptr, nullptr);
  for (int e = 0; e < ENUM; ++e) {
    moe_up<true><<<upg, 256, 0, stream>>>(xb, weg + (size_t)e * CH,
                                          weu + (size_t)e * CH, hb, cnt + e,
                                          lidx + (size_t)e * NTOK);
    moe_down<true><<<dng, 256, 0, stream>>>(hb, wed + (size_t)e * CH, out, cnt + e,
                                            lidx + (size_t)e * NTOK,
                                            lw + (size_t)e * NTOK);
  }
}
